// NodeEdge_50869592655528
// MI455X (gfx1250) — compile-verified, matrix operand order checked by reading, not run on hardware
//
#include <hip/hip_runtime.h>
#include <hip/hip_bf16.h>

#define HD 128           // feature dim (both H and D are 128)
#define TILE_ROWS 128    // rows per block (8 waves x 16)

typedef __bf16 bf16_t;
typedef __bf16 v16bf __attribute__((ext_vector_type(16)));
typedef float  v8f   __attribute__((ext_vector_type(8)));

// ---------------------------------------------------------------------------
// helpers
// ---------------------------------------------------------------------------
__device__ __forceinline__ void ld8(const float* __restrict__ p, float f[8]) {
  float4 a = ((const float4*)p)[0];
  float4 b = ((const float4*)p)[1];
  f[0] = a.x; f[1] = a.y; f[2] = a.z; f[3] = a.w;
  f[4] = b.x; f[5] = b.y; f[6] = b.z; f[7] = b.w;
}

// ---------------------------------------------------------------------------
// zero workspace regions (must run every call: graph replays accumulate)
// ---------------------------------------------------------------------------
__global__ void zero_kernel(float* __restrict__ p, long long n) {
  long long i = (long long)blockIdx.x * blockDim.x + threadIdx.x;
  long long stride = (long long)gridDim.x * blockDim.x;
  for (; i < n; i += stride) p[i] = 0.0f;
}

// ---------------------------------------------------------------------------
// pack fp32 weights [K,128] into bf16 WMMA B-operand lane layout:
//   out[((kc*8+nt)*32+lane)*16+idx],  k = kc*32 + ((idx&7)|(lanehi<<3)|((idx>>3)<<4))
// ---------------------------------------------------------------------------
__global__ void pack_w_kernel(const float* __restrict__ W, bf16_t* __restrict__ out, int K) {
  int tid = blockIdx.x * 256 + threadIdx.x;
  if (tid >= K * HD) return;
  int idx  = tid & 15;
  int lane = (tid >> 4) & 31;
  int nt   = (tid >> 9) & 7;
  int kc   = tid >> 12;
  int h    = lane >> 4;
  int kl   = (idx & 7) | (h << 3) | ((idx >> 3) << 4);
  int k    = kc * 32 + kl;
  int n    = nt * 16 + (lane & 15);
  out[tid] = (bf16_t)W[(size_t)k * HD + n];
}

// ---------------------------------------------------------------------------
// s[v] = sum of edge_attr over incident edges; degree[v] = incident count
// ---------------------------------------------------------------------------
__global__ void scatter_s_kernel(const float* __restrict__ ea, const int* __restrict__ srcI,
                                 const int* __restrict__ dstI, float* __restrict__ s,
                                 float* __restrict__ degree, int E) {
  int e = blockIdx.x * 2 + (threadIdx.x >> 7);
  int c = threadIdx.x & 127;
  if (e >= E) return;
  float v = ea[(size_t)e * HD + c];
  int a = srcI[e], b = dstI[e];
  atomicAdd(&s[(size_t)a * HD + c], v);
  atomicAdd(&s[(size_t)b * HD + c], v);
  if (c == 0) { atomicAdd(&degree[a], 1.0f); atomicAdd(&degree[b], 1.0f); }
}

// ---------------------------------------------------------------------------
// Fused GEMM + bias + BN-statistics kernel.
//   MODE 1: A = concat(edge2edge, node2edge)  (KC=8, gathers from s/x/ea)
//   MODE 2: A = relu(bn(buf))                 (KC=4)
//   MODE 3: A = (1+eps_e)*edge2edge + relu(bn(buf))   (KC=4)
//   MODE 4: A = x*((1+eps_n)-degree) + neighbor       (KC=4)
// Output: out = A@W + bias (fp32), plus per-column sum/sumsq atomics.
// ---------------------------------------------------------------------------
template <int MODE, int KC>
__global__ __launch_bounds__(256)
void gemm_bn_kernel(const v16bf* __restrict__ Wp, const float* __restrict__ bias,
                    float* __restrict__ out, float* __restrict__ gsum, float* __restrict__ gsq,
                    int R,
                    const float* __restrict__ p0, const float* __restrict__ p1,
                    const float* __restrict__ p2,
                    const int* __restrict__ srcI, const int* __restrict__ dstI,
                    const float* __restrict__ bn_scale, const float* __restrict__ bn_shift,
                    const float* __restrict__ epsp) {
  __shared__ float lsum[HD], lsq[HD], lscale[HD], lshift[HD];
  int tid = threadIdx.x;
  if (tid < HD) {
    lsum[tid] = 0.0f; lsq[tid] = 0.0f;
    if (MODE == 2 || MODE == 3) { lscale[tid] = bn_scale[tid]; lshift[tid] = bn_shift[tid]; }
  }
  __syncthreads();

  int wave    = tid >> 5;
  int lane    = tid & 31;
  int half    = lane >> 4;                      // which K-half this lane carries
  int rowbase = blockIdx.x * TILE_ROWS + wave * 16;

  if (rowbase < R) {                            // wave-uniform guard: EXEC all-1 at WMMA
    int  r  = rowbase + (lane & 15);            // A-matrix row for this lane
    bool rv = r < R;
    int  si = 0, di = 0;
    float coef = 0.0f, ce = 0.0f;
    if (MODE == 1 || MODE == 3) { if (rv) { si = srcI[r]; di = dstI[r]; } }
    if (MODE == 3) ce = 1.0f + epsp[0];
    if (MODE == 4) { if (rv) coef = (1.0f + epsp[0]) - p2[r]; }

    v8f acc[8];
    v8f z8 = {0.f, 0.f, 0.f, 0.f, 0.f, 0.f, 0.f, 0.f};
#pragma unroll
    for (int i = 0; i < 8; ++i) acc[i] = z8;

#pragma unroll
    for (int kc = 0; kc < KC; ++kc) {
      float f[16];
#pragma unroll
      for (int run = 0; run < 2; ++run) {
        int kb = kc * 32 + half * 8 + run * 16; // 8 contiguous k values
        float* fr = f + run * 8;
        if (!rv) {
#pragma unroll
          for (int j = 0; j < 8; ++j) fr[j] = 0.0f;
        } else if (MODE == 1) {
          if (kb < HD) {                        // edge2edge = s[src]+s[dst]-ea
            float a0[8], a1[8], a2[8];
            ld8(p0 + (size_t)si * HD + kb, a0);
            ld8(p0 + (size_t)di * HD + kb, a1);
            ld8(p2 + (size_t)r  * HD + kb, a2);
#pragma unroll
            for (int j = 0; j < 8; ++j) fr[j] = a0[j] + a1[j] - a2[j];
          } else {                              // node2edge = x[src]+x[dst]
            int kb2 = kb - HD;
            float a0[8], a1[8];
            ld8(p1 + (size_t)si * HD + kb2, a0);
            ld8(p1 + (size_t)di * HD + kb2, a1);
#pragma unroll
            for (int j = 0; j < 8; ++j) fr[j] = a0[j] + a1[j];
          }
        } else if (MODE == 2) {
          float a0[8];
          ld8(p0 + (size_t)r * HD + kb, a0);
#pragma unroll
          for (int j = 0; j < 8; ++j)
            fr[j] = fmaxf(a0[j] * lscale[kb + j] + lshift[kb + j], 0.0f);
        } else if (MODE == 3) {
          float a0[8], a1[8], a2[8], a3[8];
          ld8(p0 + (size_t)si * HD + kb, a0);   // s[src]
          ld8(p0 + (size_t)di * HD + kb, a1);   // s[dst]
          ld8(p2 + (size_t)r  * HD + kb, a2);   // edge_attr
          ld8(p1 + (size_t)r  * HD + kb, a3);   // eo pre-BN
#pragma unroll
          for (int j = 0; j < 8; ++j) {
            float eo = fmaxf(a3[j] * lscale[kb + j] + lshift[kb + j], 0.0f);
            fr[j] = ce * (a0[j] + a1[j] - a2[j]) + eo;
          }
        } else {                                // MODE 4
          float a0[8], a1[8];
          ld8(p0 + (size_t)r * HD + kb, a0);    // x
          ld8(p1 + (size_t)r * HD + kb, a1);    // neighbor
#pragma unroll
          for (int j = 0; j < 8; ++j) fr[j] = a0[j] * coef + a1[j];
        }
      }
      v16bf av;
#pragma unroll
      for (int j = 0; j < 16; ++j) av[j] = (bf16_t)f[j];

      const v16bf* wb = Wp + (size_t)kc * 256 + lane;
#pragma unroll
      for (int nt = 0; nt < 8; ++nt) {
        v16bf bv = wb[nt * 32];
        acc[nt] = __builtin_amdgcn_wmma_f32_16x16x32_bf16(
            false, av, false, bv, (short)0, acc[nt], false, false);
      }
    }

    // epilogue: bias + fp32 store + per-column BN partial stats into LDS
    int mbase = rowbase + half * 8;             // C layout: lane half selects M 0-7 / 8-15
#pragma unroll
    for (int nt = 0; nt < 8; ++nt) {
      int col = nt * 16 + (lane & 15);
      float bia = bias[col];
      float ps = 0.0f, pq = 0.0f;
#pragma unroll
      for (int i = 0; i < 8; ++i) {
        int row = mbase + i;
        if (row < R) {
          float v = acc[nt][i] + bia;
          out[(size_t)row * HD + col] = v;
          ps += v; pq += v * v;
        }
      }
      atomicAdd(&lsum[col], ps);
      atomicAdd(&lsq[col], pq);
    }
  }
  __syncthreads();
  if (tid < HD) {
    atomicAdd(&gsum[tid], lsum[tid]);
    atomicAdd(&gsq[tid], lsq[tid]);
  }
}

// ---------------------------------------------------------------------------
// BN statistics -> per-column affine (scale/shift), biased variance
// ---------------------------------------------------------------------------
__global__ void bn_finalize_kernel(const float* __restrict__ gsum, const float* __restrict__ gsq,
                                   const float* __restrict__ gamma, const float* __restrict__ beta,
                                   float* __restrict__ scale, float* __restrict__ shift,
                                   float invR) {
  int c = threadIdx.x;
  float m  = gsum[c] * invR;
  float v  = gsq[c] * invR - m * m;
  float rs = rsqrtf(v + 1e-5f);
  float sc = gamma[c] * rs;
  scale[c] = sc;
  shift[c] = beta[c] - m * sc;
}

// ---------------------------------------------------------------------------
// final edge BN+ReLU apply, write edge_out, scatter into neighbor
// ---------------------------------------------------------------------------
__global__ void apply_scatter_kernel(const float* __restrict__ buf,
                                     const float* __restrict__ scale, const float* __restrict__ shift,
                                     const int* __restrict__ srcI, const int* __restrict__ dstI,
                                     float* __restrict__ edge_out, float* __restrict__ neighbor,
                                     int E) {
  int e = blockIdx.x * 2 + (threadIdx.x >> 7);
  int c = threadIdx.x & 127;
  if (e >= E) return;
  float v = fmaxf(buf[(size_t)e * HD + c] * scale[c] + shift[c], 0.0f);
  edge_out[(size_t)e * HD + c] = v;
  atomicAdd(&neighbor[(size_t)srcI[e] * HD + c], v);
  atomicAdd(&neighbor[(size_t)dstI[e] * HD + c], v);
}

// ---------------------------------------------------------------------------
// final node BN+ReLU apply
// ---------------------------------------------------------------------------
__global__ void apply_node_kernel(const float* __restrict__ buf,
                                  const float* __restrict__ scale, const float* __restrict__ shift,
                                  float* __restrict__ outp, int N) {
  int r = blockIdx.x * 2 + (threadIdx.x >> 7);
  int c = threadIdx.x & 127;
  if (r >= N) return;
  outp[(size_t)r * HD + c] = fmaxf(buf[(size_t)r * HD + c] * scale[c] + shift[c], 0.0f);
}

// ---------------------------------------------------------------------------
extern "C" void kernel_launch(void* const* d_in, const int* in_sizes, int n_in,
                              void* d_out, int out_size, void* d_ws, size_t ws_size,
                              hipStream_t stream) {
  const float* x    = (const float*)d_in[0];
  const float* ea   = (const float*)d_in[1];
  const int*   eidx = (const int*)d_in[2];
  // node MLP params
  const float* nW1 = (const float*)d_in[3];  const float* nb1 = (const float*)d_in[4];
  const float* ng1 = (const float*)d_in[5];  const float* nbe1= (const float*)d_in[6];
  const float* nW2 = (const float*)d_in[7];  const float* nb2 = (const float*)d_in[8];
  const float* ng2 = (const float*)d_in[9];  const float* nbe2= (const float*)d_in[10];
  // edge MLP1 params
  const float* e1W1 = (const float*)d_in[11]; const float* e1b1 = (const float*)d_in[12];
  const float* e1g1 = (const float*)d_in[13]; const float* e1be1= (const float*)d_in[14];
  const float* e1W2 = (const float*)d_in[15]; const float* e1b2 = (const float*)d_in[16];
  const float* e1g2 = (const float*)d_in[17]; const float* e1be2= (const float*)d_in[18];
  // edge MLP2 params
  const float* e2W1 = (const float*)d_in[19]; const float* e2b1 = (const float*)d_in[20];
  const float* e2g1 = (const float*)d_in[21]; const float* e2be1= (const float*)d_in[22];
  const float* e2W2 = (const float*)d_in[23]; const float* e2b2 = (const float*)d_in[24];
  const float* e2g2 = (const float*)d_in[25]; const float* e2be2= (const float*)d_in[26];
  const float* eps_node = (const float*)d_in[27];
  const float* eps_edge = (const float*)d_in[28];

  int N = in_sizes[0] / HD;
  int E = in_sizes[1] / HD;

  // workspace layout
  float* wsf = (float*)d_ws;
  size_t off = 0;
  float* s_buf  = wsf + off; off += (size_t)N * HD;          // s[v]; reused as neighbor later
  float* degree = wsf + off; off += (size_t)((N + 31) / 32) * 32;
  float* stats  = wsf + off; off += 6 * 256;                 // per stage: sum[128], sumsq[128]
  float* affine = wsf + off; off += 6 * 256;                 // per stage: scale[128], shift[128]
  size_t zeroCount = off;
  float* bufA = wsf + off; off += (size_t)E * HD;
  float* bufB = wsf + off; off += (size_t)E * HD;
  bf16_t* pw     = (bf16_t*)(wsf + off);
  bf16_t* pw_e1a = pw;                  // K=256 -> 8 kchunks
  bf16_t* pw_e1b = pw_e1a + 8 * 4096;
  bf16_t* pw_e2a = pw_e1b + 4 * 4096;
  bf16_t* pw_e2b = pw_e2a + 4 * 4096;
  bf16_t* pw_na  = pw_e2b + 4 * 4096;
  bf16_t* pw_nb  = pw_na  + 4 * 4096;

  const int* srcI = eidx;
  const int* dstI = eidx + E;
  float* neighbor = s_buf;              // reuse: s dead after G3, zeroed before scatter

  float* node_out = (float*)d_out;
  float* edge_out = node_out + (size_t)N * HD;

  float* SUM[6], *SQ[6], *SC[6], *SH[6];
  for (int i = 0; i < 6; ++i) {
    SUM[i] = stats + i * 256; SQ[i] = SUM[i] + 128;
    SC[i]  = affine + i * 256; SH[i] = SC[i] + 128;
  }

  int eg = (E + TILE_ROWS - 1) / TILE_ROWS;
  int ng = (N + TILE_ROWS - 1) / TILE_ROWS;
  float invE = 1.0f / (float)E, invN = 1.0f / (float)N;

  // 0) re-zero accumulators (graph replays must be self-contained)
  zero_kernel<<<2048, 256, 0, stream>>>(wsf, (long long)zeroCount);
  // 1) pack weights into WMMA bf16 B layout
  pack_w_kernel<<<(256 * HD + 255) / 256, 256, 0, stream>>>(e1W1, pw_e1a, 256);
  pack_w_kernel<<<(128 * HD + 255) / 256, 256, 0, stream>>>(e1W2, pw_e1b, 128);
  pack_w_kernel<<<(128 * HD + 255) / 256, 256, 0, stream>>>(e2W1, pw_e2a, 128);
  pack_w_kernel<<<(128 * HD + 255) / 256, 256, 0, stream>>>(e2W2, pw_e2b, 128);
  pack_w_kernel<<<(128 * HD + 255) / 256, 256, 0, stream>>>(nW1,  pw_na,  128);
  pack_w_kernel<<<(128 * HD + 255) / 256, 256, 0, stream>>>(nW2,  pw_nb,  128);
  // 2) s[v] and degree[v]
  scatter_s_kernel<<<(E + 1) / 2, 256, 0, stream>>>(ea, srcI, dstI, s_buf, degree, E);
  // 3) edge MLP1 layer 1: fused concat GEMM (K=256)
  gemm_bn_kernel<1, 8><<<eg, 256, 0, stream>>>((const v16bf*)pw_e1a, e1b1, bufA, SUM[0], SQ[0], E,
      s_buf, x, ea, srcI, dstI, nullptr, nullptr, nullptr);
  bn_finalize_kernel<<<1, 128, 0, stream>>>(SUM[0], SQ[0], e1g1, e1be1, SC[0], SH[0], invE);
  // 4) edge MLP1 layer 2
  gemm_bn_kernel<2, 4><<<eg, 256, 0, stream>>>((const v16bf*)pw_e1b, e1b2, bufB, SUM[1], SQ[1], E,
      bufA, nullptr, nullptr, nullptr, nullptr, SC[0], SH[0], nullptr);
  bn_finalize_kernel<<<1, 128, 0, stream>>>(SUM[1], SQ[1], e1g2, e1be2, SC[1], SH[1], invE);
  // 5) edge MLP2 layer 1: (1+eps_e)*edge2edge + relu(bn(eo))
  gemm_bn_kernel<3, 4><<<eg, 256, 0, stream>>>((const v16bf*)pw_e2a, e2b1, bufA, SUM[2], SQ[2], E,
      s_buf, bufB, ea, srcI, dstI, SC[1], SH[1], eps_edge);
  bn_finalize_kernel<<<1, 128, 0, stream>>>(SUM[2], SQ[2], e2g1, e2be1, SC[2], SH[2], invE);
  // 6) edge MLP2 layer 2
  gemm_bn_kernel<2, 4><<<eg, 256, 0, stream>>>((const v16bf*)pw_e2b, e2b2, bufB, SUM[3], SQ[3], E,
      bufA, nullptr, nullptr, nullptr, nullptr, SC[2], SH[2], nullptr);
  bn_finalize_kernel<<<1, 128, 0, stream>>>(SUM[3], SQ[3], e2g2, e2be2, SC[3], SH[3], invE);
  // 7) edge_out apply + neighbor scatter (s buffer recycled)
  zero_kernel<<<2048, 256, 0, stream>>>(neighbor, (long long)N * HD);
  apply_scatter_kernel<<<(E + 1) / 2, 256, 0, stream>>>(bufB, SC[3], SH[3], srcI, dstI,
                                                        edge_out, neighbor, E);
  // 8) node MLP layer 1: x*((1+eps_n)-degree) + neighbor
  gemm_bn_kernel<4, 4><<<ng, 256, 0, stream>>>((const v16bf*)pw_na, nb1, bufA, SUM[4], SQ[4], N,
      x, neighbor, degree, nullptr, nullptr, nullptr, nullptr, eps_node);
  bn_finalize_kernel<<<1, 128, 0, stream>>>(SUM[4], SQ[4], ng1, nbe1, SC[4], SH[4], invN);
  // 9) node MLP layer 2
  gemm_bn_kernel<2, 4><<<ng, 256, 0, stream>>>((const v16bf*)pw_nb, nb2, bufB, SUM[5], SQ[5], N,
      bufA, nullptr, nullptr, nullptr, nullptr, SC[4], SH[4], nullptr);
  bn_finalize_kernel<<<1, 128, 0, stream>>>(SUM[5], SQ[5], ng2, nbe2, SC[5], SH[5], invN);
  // 10) node_out apply
  apply_node_kernel<<<(N + 1) / 2, 256, 0, stream>>>(bufB, SC[5], SH[5], node_out, N);
}
